// SelfAttention_86242943304257
// MI455X (gfx1250) — matrix operand order repaired and checked by reading, not compiled, not run on hardware
//
#include <hip/hip_runtime.h>
#include <hip/hip_bf16.h>

#define L_SEQ 2048
#define BATCH 2
#define HID 1024
#define HEADS 16
#define DH 64
#define TOKENS (L_SEQ * BATCH)   // 4096
#define SCALE_F 0.125f           // 64^-0.5

typedef __attribute__((ext_vector_type(16))) __bf16 bf16x16;
typedef __attribute__((ext_vector_type(8)))  float  floatx8;
typedef __attribute__((ext_vector_type(4)))  unsigned int uint32x4;
typedef __attribute__((ext_vector_type(8)))  int int32x8;
typedef __attribute__((ext_vector_type(4)))  int int32x4;

union Frag { bf16x16 v; unsigned int u[8]; };

__device__ __forceinline__ unsigned short f2bf(float x) {
    union { float f; unsigned int u; } c; c.f = x;
    unsigned int r = ((c.u >> 16) & 1u) + 0x7FFFu;   // round-to-nearest-even
    return (unsigned short)((c.u + r) >> 16);
}
__device__ __forceinline__ unsigned int packbf2(float a, float b) {
#if __has_builtin(__builtin_amdgcn_cvt_pk_bf16_f32)
    auto pk = __builtin_amdgcn_cvt_pk_bf16_f32(a, b);
    unsigned int u; __builtin_memcpy(&u, &pk, sizeof(u));
    return u;
#else
    return (unsigned int)f2bf(a) | ((unsigned int)f2bf(b) << 16);
#endif
}
__device__ __forceinline__ floatx8 zero8() {
    floatx8 z = {0.f, 0.f, 0.f, 0.f, 0.f, 0.f, 0.f, 0.f};
    return z;
}

// Async DMA: 16 bytes global -> LDS, tracked by ASYNCcnt (no VGPR data).
__device__ __forceinline__ void async_copy_b128(const unsigned short* gsrc,
                                                unsigned short* ldst) {
    unsigned loff = (unsigned)(size_t)ldst;   // LDS aperture: addr[31:0] = LDS offset
    asm volatile("global_load_async_to_lds_b128 %0, %1, off"
                 :: "v"(loff), "v"(gsrc) : "memory");
}
__device__ __forceinline__ void wait_async0() {
    asm volatile("s_wait_asynccnt 0" ::: "memory");
}

// --------------------------- TDM descriptor helpers ------------------------
// D# group1 flags: data_size=2B, pad_enable, pad_interval=4 (every 32 DWORDs =
// one 128 B tile row), pad_amount=11 (12 DWORDs = 48 B) -> LDS pitch 176 B.
#define TDM_FLAGS ((1u << 16) | (1u << 20) | (4u << 22) | (11u << 25))

__device__ __forceinline__ uint32x4 tdm_g0(unsigned lds_addr,
                                           unsigned long long gaddr) {
    uint32x4 g0 = { 1u,                                   // count=1, user mode
                    lds_addr,                             // [63:32]  lds_addr
                    (unsigned)(gaddr & 0xffffffffu),      // [95:64]  addr lo
                    (unsigned)((gaddr >> 32) & 0x01ffffffu) | (2u << 30) }; // type=2
    return g0;
}
// 2-D tile: tensor [td1 rows x td0 elems] stride s0; tile t1 x t0 elems.
__device__ __forceinline__ int32x8 tdm_g1(unsigned td0, unsigned td1,
                                          unsigned t0, unsigned t1,
                                          unsigned long long s0) {
    int32x8 g;
    g[0] = (int)TDM_FLAGS;
    g[1] = (int)((td0 & 0xffffu) << 16);                       // abar=0 | td0 lo
    g[2] = (int)((td0 >> 16) | ((td1 & 0xffffu) << 16));
    g[3] = (int)((td1 >> 16) | ((t0 & 0xffffu) << 16));        // tile_dim0
    g[4] = (int)(t1 & 0xffffu);                                // tile_dim1, dim2=0
    g[5] = (int)(unsigned)(s0 & 0xffffffffu);                  // dim0_stride lo
    g[6] = (int)(unsigned)((s0 >> 32) & 0xffffu);              // stride hi | s1 lo=0
    g[7] = 0;                                                  // dim1_stride=0 (2D)
    return g;
}

// ---------------------------------------------------------------------------
// Kernel 1: QKV projection.  y = x @ W^T + b  (fp32 in, bf16 out)
// blockIdx.z selects Q / K / V.  Q is pre-scaled by 1/sqrt(dh).
// Q,K written [B,H,L,dh]; V written transposed [B,H,dh,L].
// ---------------------------------------------------------------------------
#define PBM 64
#define PBN 128
#define PBK 32
#define PPITCH 40   // bf16 elems per LDS row (80 B: 16-aligned, padded)

__global__ __launch_bounds__(256)
void qkv_proj_kernel(const float* __restrict__ xq, const float* __restrict__ xk,
                     const float* __restrict__ xv,
                     const float* __restrict__ wq, const float* __restrict__ wk,
                     const float* __restrict__ wv,
                     const float* __restrict__ bq, const float* __restrict__ bk,
                     const float* __restrict__ bv,
                     unsigned short* __restrict__ oq, unsigned short* __restrict__ ok,
                     unsigned short* __restrict__ ovt)
{
    __shared__ __align__(16) unsigned short lA[PBM * PPITCH];
    __shared__ __align__(16) unsigned short lB[PBN * PPITCH];

    const int which = blockIdx.z;
    const float* __restrict__ x    = (which == 0) ? xq : (which == 1) ? xk : xv;
    const float* __restrict__ w    = (which == 0) ? wq : (which == 1) ? wk : wv;
    const float* __restrict__ bias = (which == 0) ? bq : (which == 1) ? bk : bv;

    const int t0   = blockIdx.x * PBM;   // first token (row = l*B+b of [L,B,HID])
    const int n0   = blockIdx.y * PBN;   // first output feature
    const int tid  = threadIdx.x;
    const int lane = tid & 31;
    const int lm   = lane & 15;
    const int lh   = lane >> 4;
    const int wave = tid >> 5;
    const int wr   = wave & 3;           // token sub-tile (16 rows)
    const int wc   = wave >> 2;          // feature sub-tile (64 cols)

    floatx8 acc[4] = {zero8(), zero8(), zero8(), zero8()};

    for (int kk = 0; kk < HID; kk += PBK) {
        if (kk + PBK < HID)   // speculative prefetch of next weight K-slab
            __builtin_prefetch(w + (size_t)(n0 + (tid & 127)) * HID + kk + PBK, 0, 0);
        #pragma unroll
        for (int p = 0; p < 4; ++p) {    // A: 64x32 fp32 -> bf16 pairs
            int idx = tid + p * 256;
            int row = idx >> 4, cp = idx & 15;
            const float* s = x + (size_t)(t0 + row) * HID + kk + cp * 2;
            *(unsigned int*)&lA[row * PPITCH + cp * 2] = packbf2(s[0], s[1]);
        }
        #pragma unroll
        for (int p = 0; p < 8; ++p) {    // B: 128x32 rows of W (N x K row-major)
            int idx = tid + p * 256;
            int row = idx >> 4, cp = idx & 15;
            const float* s = w + (size_t)(n0 + row) * HID + kk + cp * 2;
            *(unsigned int*)&lB[row * PPITCH + cp * 2] = packbf2(s[0], s[1]);
        }
        __syncthreads();

        Frag a;   // 16x32 A fragment (ISA 7.12.2 16-bit A layout)
        #pragma unroll
        for (int v = 0; v < 8; ++v) {
            int k = (v >> 2) * 16 + lh * 8 + (v & 3) * 2;
            a.u[v] = *(const unsigned int*)&lA[(wr * 16 + lm) * PPITCH + k];
        }
        #pragma unroll
        for (int t = 0; t < 4; ++t) {
            Frag b;   // 32x16 B fragment from N x K rows
            #pragma unroll
            for (int v = 0; v < 8; ++v) {
                int k = (v >> 2) * 16 + lh * 8 + (v & 3) * 2;
                b.u[v] = *(const unsigned int*)&lB[(wc * 64 + t * 16 + lm) * PPITCH + k];
            }
            acc[t] = __builtin_amdgcn_wmma_f32_16x16x32_bf16(
                false, a.v, false, b.v, (short)0, acc[t], false, false);
        }
        __syncthreads();
    }

    const float qscale = (which == 0) ? SCALE_F : 1.0f;
    #pragma unroll
    for (int t = 0; t < 4; ++t) {
        const int ng    = n0 + wc * 64 + t * 16 + lm;  // C layout: lane = N
        const float bv2 = bias[ng];
        const int h  = ng >> 6;
        const int dh = ng & 63;
        #pragma unroll
        for (int r = 0; r < 8; ++r) {                  // C layout: M = lh*8 + r
            const int tok = t0 + wr * 16 + lh * 8 + r;
            const int l   = tok >> 1;                  // B == 2
            const int bb  = tok & 1;
            const unsigned short val = f2bf((acc[t][r] + bv2) * qscale);
            if (which == 2)
                ovt[(((size_t)bb * HEADS + h) * DH + dh) * L_SEQ + l] = val;
            else if (which == 0)
                oq[(((size_t)bb * HEADS + h) * L_SEQ + l) * DH + dh] = val;
            else
                ok[(((size_t)bb * HEADS + h) * L_SEQ + l) * DH + dh] = val;
        }
    }
}

// ---------------------------------------------------------------------------
// Kernel 2: flash attention with Gaussian positional bias + key mask.
// One block = 128 queries of one (b,h); 8 waves x 16 query rows; K/V tiles
// streamed by the Tensor Data Mover (wave 0 issues, TENSORcnt tracked, LDS
// row padding 128B+48B giving the 176 B conflict-free pitch).
// ---------------------------------------------------------------------------
#define AQW 8
#define AQB (AQW * 16)   // 128 queries / block
#define AKB 64           // keys per iteration
#define KP  88           // LDS pitch: 176 B (16-aligned, conflict-free frag rows)
#define PP  68           // LDS pitch for per-wave P tiles

__global__ __launch_bounds__(256)
void attention_kernel(const unsigned short* __restrict__ Q,
                      const unsigned short* __restrict__ K,
                      const unsigned short* __restrict__ Vt,
                      const unsigned char* __restrict__ mask,
                      const float* __restrict__ tao,
                      unsigned short* __restrict__ X)
{
    __shared__ __align__(16) unsigned short lK[AKB * KP];        // [key][dh]
    __shared__ __align__(16) unsigned short lV[DH * KP];         // [dh][key]
    __shared__ __align__(16) unsigned short lP[AQW * 16 * PP];   // P transpose bounce
    __shared__ float lMB[AKB];

    const int bh   = blockIdx.y;
    const int b    = bh >> 4;       // / HEADS
    const int h    = bh & 15;
    const int q0   = blockIdx.x * AQB;
    const int tid  = threadIdx.x;
    const int lane = tid & 31;
    const int lm   = lane & 15;
    const int lh   = lane >> 4;
    const int wave = tid >> 5;

    const float tv    = tao[h];
    const float itao4 = 1.0f / (tv * tv * tv * tv);

    // TDM descriptors (group1 is loop-invariant; group0 global addr moves)
    const int32x8 g1K = tdm_g1(/*td0=*/DH,    /*td1=*/L_SEQ, /*t0=*/DH,  /*t1=*/AKB, /*s0=*/DH);
    const int32x8 g1V = tdm_g1(/*td0=*/L_SEQ, /*td1=*/DH,    /*t0=*/AKB, /*t1=*/DH,  /*s0=*/L_SEQ);
    const int32x4 gz4 = {0, 0, 0, 0};
    const int32x8 gz8 = {0, 0, 0, 0, 0, 0, 0, 0};
    const unsigned ldsK = (unsigned)(size_t)&lK[0];
    const unsigned ldsV = (unsigned)(size_t)&lV[0];

    // Q fragments for this wave's 16 rows, 2 k-chunks of 32 dh (loaded once)
    Frag qf[2];
    {
        const size_t qbase = ((size_t)bh * L_SEQ + (q0 + wave * 16 + lm)) * DH;
        #pragma unroll
        for (int c = 0; c < 2; ++c)
            #pragma unroll
            for (int v = 0; v < 8; ++v) {
                int k = c * 32 + (v >> 2) * 16 + lh * 8 + (v & 3) * 2;
                qf[c].u[v] = *(const unsigned int*)(Q + qbase + k);
            }
    }

    floatx8 o[4] = {zero8(), zero8(), zero8(), zero8()};
    float rmax[8], rsum[8];
    #pragma unroll
    for (int r = 0; r < 8; ++r) { rmax[r] = -3.0e38f; rsum[r] = 0.f; }

    for (int j0 = 0; j0 < L_SEQ; j0 += AKB) {
        // TDM: wave 0 DMAs both tiles (EXEC-independent; 1 wave avoids dup DMAs)
        if (wave == 0) {
            unsigned long long gK = (unsigned long long)(size_t)
                (K + ((size_t)bh * L_SEQ + j0) * DH);
            unsigned long long gV = (unsigned long long)(size_t)
                (Vt + (size_t)bh * DH * L_SEQ + j0);
            __builtin_amdgcn_tensor_load_to_lds(tdm_g0(ldsK, gK), g1K, gz4, gz4, gz8, 0);
            __builtin_amdgcn_tensor_load_to_lds(tdm_g0(ldsV, gV), g1V, gz4, gz4, gz8, 0);
        }
        if (tid < AKB)
            lMB[tid] = mask[(size_t)b * L_SEQ + j0 + tid] ? -__builtin_inff() : 0.0f;
        if (wave == 0)
            __builtin_amdgcn_s_wait_tensorcnt(0);   // drain before signaling
        __syncthreads();

        // S = Q K^T (16 x 64 per wave)
        floatx8 s[4];
        #pragma unroll
        for (int t = 0; t < 4; ++t) {
            s[t] = zero8();
            #pragma unroll
            for (int c = 0; c < 2; ++c) {
                Frag kf;
                #pragma unroll
                for (int v = 0; v < 8; ++v)
                    kf.u[v] = *(const unsigned int*)
                        &lK[(t * 16 + lm) * KP + c * 32 + (v >> 2) * 16 + lh * 8 + (v & 3) * 2];
                s[t] = __builtin_amdgcn_wmma_f32_16x16x32_bf16(
                    false, qf[c].v, false, kf.v, (short)0, s[t], false, false);
            }
        }

        // Gaussian bias + mask, then row max across the 64 keys
        const int ibase = q0 + wave * 16 + lh * 8;
        float tmax[8];
        #pragma unroll
        for (int r = 0; r < 8; ++r) tmax[r] = -3.0e38f;
        #pragma unroll
        for (int t = 0; t < 4; ++t) {
            const int j    = j0 + t * 16 + lm;
            const float mb = lMB[t * 16 + lm];
            #pragma unroll
            for (int r = 0; r < 8; ++r) {
                float d   = (float)(ibase + r - j);
                float val = s[t][r] - 0.5f * d * d * itao4 + mb;
                s[t][r]   = val;
                tmax[r]   = fmaxf(tmax[r], val);
            }
        }
        #pragma unroll
        for (int off = 1; off < 16; off <<= 1)
            #pragma unroll
            for (int r = 0; r < 8; ++r)
                tmax[r] = fmaxf(tmax[r], __shfl_xor(tmax[r], off, 32));

        float alpha[8];
        #pragma unroll
        for (int r = 0; r < 8; ++r) {
            float mn = fmaxf(rmax[r], tmax[r]);
            alpha[r] = __expf(rmax[r] - mn);
            rmax[r]  = mn;
        }

        // P = exp(S - m); bf16 to per-wave LDS (C-layout -> A-layout transpose)
        float psum[8];
        #pragma unroll
        for (int r = 0; r < 8; ++r) psum[r] = 0.f;
        #pragma unroll
        for (int t = 0; t < 4; ++t)
            #pragma unroll
            for (int r = 0; r < 8; ++r) {
                float p = __expf(s[t][r] - rmax[r]);
                psum[r] += p;
                lP[(wave * 16 + lh * 8 + r) * PP + t * 16 + lm] = f2bf(p);
            }
        #pragma unroll
        for (int off = 1; off < 16; off <<= 1)
            #pragma unroll
            for (int r = 0; r < 8; ++r)
                psum[r] += __shfl_xor(psum[r], off, 32);
        #pragma unroll
        for (int r = 0; r < 8; ++r) rsum[r] = rsum[r] * alpha[r] + psum[r];

        #pragma unroll
        for (int t = 0; t < 4; ++t)
            #pragma unroll
            for (int r = 0; r < 8; ++r)
                o[t][r] *= alpha[r];

        asm volatile("s_wait_dscnt 0" ::: "memory");  // per-wave LDS RAW on lP

        // O += P @ V
        Frag pf[2];
        #pragma unroll
        for (int c = 0; c < 2; ++c)
            #pragma unroll
            for (int v = 0; v < 8; ++v) {
                int k = c * 32 + (v >> 2) * 16 + lh * 8 + (v & 3) * 2;
                pf[c].u[v] = *(const unsigned int*)&lP[(wave * 16 + lm) * PP + k];
            }
        #pragma unroll
        for (int t = 0; t < 4; ++t)
            #pragma unroll
            for (int c = 0; c < 2; ++c) {
                Frag vf;
                #pragma unroll
                for (int v = 0; v < 8; ++v)
                    vf.u[v] = *(const unsigned int*)
                        &lV[(t * 16 + lm) * KP + c * 32 + (v >> 2) * 16 + lh * 8 + (v & 3) * 2];
                o[t] = __builtin_amdgcn_wmma_f32_16x16x32_bf16(
                    false, pf[c].v, false, vf.v, (short)0, o[t], false, false);
            }
        __syncthreads();
    }

    // normalize and store X bf16 [token][HID], token = l*B + b
    #pragma unroll
    for (int r = 0; r < 8; ++r) {
        const int q      = q0 + wave * 16 + lh * 8 + r;
        const size_t tok = (size_t)q * BATCH + b;
        const float rinv = 1.0f / rsum[r];
        #pragma unroll
        for (int t = 0; t < 4; ++t)
            X[tok * HID + h * DH + t * 16 + lm] = f2bf(o[t][r] * rinv);
    }
}

// ---------------------------------------------------------------------------
// Kernel 3: output projection.  out = X @ wfc^T + bfc, fp32 out in [L,B,HID]
// ---------------------------------------------------------------------------
__global__ __launch_bounds__(256)
void out_proj_kernel(const unsigned short* __restrict__ X,
                     const float* __restrict__ wfc, const float* __restrict__ bfc,
                     float* __restrict__ out)
{
    __shared__ __align__(16) unsigned short lA[PBM * PPITCH];
    __shared__ __align__(16) unsigned short lB[PBN * PPITCH];

    const int t0   = blockIdx.x * PBM;
    const int n0   = blockIdx.y * PBN;
    const int tid  = threadIdx.x;
    const int lane = tid & 31;
    const int lm   = lane & 15;
    const int lh   = lane >> 4;
    const int wave = tid >> 5;
    const int wr   = wave & 3;
    const int wc   = wave >> 2;

    floatx8 acc[4] = {zero8(), zero8(), zero8(), zero8()};

    for (int kk = 0; kk < HID; kk += PBK) {
        if (kk + PBK < HID)
            __builtin_prefetch(wfc + (size_t)(n0 + (tid & 127)) * HID + kk + PBK, 0, 0);
        {   // A: already bf16 -> async DMA, 64 rows x 4 16B-chunks, 1 per thread
            int row = tid >> 2, cp = tid & 3;
            async_copy_b128(X + (size_t)(t0 + row) * HID + kk + cp * 8,
                            &lA[row * PPITCH + cp * 8]);
        }
        #pragma unroll
        for (int p = 0; p < 8; ++p) {   // B: fp32 weights -> bf16
            int idx = tid + p * 256;
            int row = idx >> 4, cp = idx & 15;
            const float* s = wfc + (size_t)(n0 + row) * HID + kk + cp * 2;
            *(unsigned int*)&lB[row * PPITCH + cp * 2] = packbf2(s[0], s[1]);
        }
        wait_async0();
        __syncthreads();

        Frag a;
        #pragma unroll
        for (int v = 0; v < 8; ++v) {
            int k = (v >> 2) * 16 + lh * 8 + (v & 3) * 2;
            a.u[v] = *(const unsigned int*)&lA[(wr * 16 + lm) * PPITCH + k];
        }
        #pragma unroll
        for (int t = 0; t < 4; ++t) {
            Frag bfr;
            #pragma unroll
            for (int v = 0; v < 8; ++v) {
                int k = (v >> 2) * 16 + lh * 8 + (v & 3) * 2;
                bfr.u[v] = *(const unsigned int*)&lB[(wc * 64 + t * 16 + lm) * PPITCH + k];
            }
            acc[t] = __builtin_amdgcn_wmma_f32_16x16x32_bf16(
                false, a.v, false, bfr.v, (short)0, acc[t], false, false);
        }
        __syncthreads();
    }

    #pragma unroll
    for (int t = 0; t < 4; ++t) {
        const int ng    = n0 + wc * 64 + t * 16 + lm;
        const float bv2 = bfc[ng];
        #pragma unroll
        for (int r = 0; r < 8; ++r) {
            const int tok = t0 + wr * 16 + lh * 8 + r;   // == [L,B] flattened
            out[(size_t)tok * HID + ng] = acc[t][r] + bv2;
        }
    }
}

// ---------------------------------------------------------------------------
extern "C" void kernel_launch(void* const* d_in, const int* in_sizes, int n_in,
                              void* d_out, int out_size, void* d_ws, size_t ws_size,
                              hipStream_t stream)
{
    (void)in_sizes; (void)n_in; (void)out_size; (void)ws_size;
    const float* q           = (const float*)d_in[0];
    const float* k           = (const float*)d_in[1];
    const float* v           = (const float*)d_in[2];
    const unsigned char* msk = (const unsigned char*)d_in[3];  // jax bool = 1 byte
    const float* wq  = (const float*)d_in[4];
    const float* wk  = (const float*)d_in[5];
    const float* wv  = (const float*)d_in[6];
    const float* wfc = (const float*)d_in[7];
    const float* bq  = (const float*)d_in[8];
    const float* bk  = (const float*)d_in[9];
    const float* bv  = (const float*)d_in[10];
    const float* bfc = (const float*)d_in[11];
    const float* tao = (const float*)d_in[12];

    const size_t QSZ = (size_t)BATCH * HEADS * L_SEQ * DH;   // 4M bf16 elems each
    unsigned short* Qbf = (unsigned short*)d_ws;
    unsigned short* Kbf = Qbf + QSZ;
    unsigned short* Vtb = Kbf + QSZ;
    unsigned short* Xbf = Vtb + QSZ;                          // total 32 MB

    qkv_proj_kernel<<<dim3(TOKENS / PBM, HID / PBN, 3), 256, 0, stream>>>(
        q, k, v, wq, wk, wv, bq, bk, bv, Qbf, Kbf, Vtb);
    attention_kernel<<<dim3(L_SEQ / AQB, BATCH * HEADS), 256, 0, stream>>>(
        Qbf, Kbf, Vtb, msk, tao, Xbf);
    out_proj_kernel<<<dim3(TOKENS / PBM, HID / PBN), 256, 0, stream>>>(
        Xbf, wfc, bfc, (float*)d_out);
}
